// DecisionTransformer_3092376453743
// MI455X (gfx1250) — matrix operand order repaired and checked by reading, not compiled, not run on hardware
//
#include <hip/hip_runtime.h>
#include <hip/hip_bf16.h>

#define BB 32
#define TT 256
#define EE 512
#define HH 4
#define HD 128
#define OBS 128
#define ACTN 18
#define NBLK 8
#define SS (3 * TT)        // 768
#define FFD (4 * EE)       // 2048
#define NTOK (BB * SS)     // 24576
#define QS (3 * EE)        // packed QKV row stride (1536)
#define VTS 40             // padded V^T LDS row stride (bf16 elems), 80B = 16B-aligned

typedef __bf16 bf16;
typedef __attribute__((ext_vector_type(16))) __bf16 v16bf;
typedef __attribute__((ext_vector_type(8)))  __bf16 v8bf;
typedef __attribute__((ext_vector_type(4)))  __bf16 v4bf;
typedef __attribute__((ext_vector_type(8)))  float  v8f;

// ---- cross-lane helpers ----------------------------------------------------
// ds_swizzle group-of-32 xor pattern: offset = (xor<<10) | and_mask(0x1f)
template <int OFF>
static __device__ inline float swz(float x) {
  return __int_as_float(__builtin_amdgcn_ds_swizzle(__float_as_int(x), OFF));
}
#define SWZ_X1  0x041F
#define SWZ_X2  0x081F
#define SWZ_X4  0x101F
#define SWZ_X8  0x201F
#define SWZ_X16 0x401F

// ---- fragment loaders ------------------------------------------------------
// A-frag (16x32 bf16, M x K): lane m holds k = e + half*16
// -> two contiguous 16B chunks at +0 and +8 elements (pointer pre-offset by half*16).
static __device__ inline v16bf load_a_frag(const bf16* p) {
  union { v16bf v; v8bf h[2]; } u;
  u.h[0] = *(const v8bf*)(p);
  u.h[1] = *(const v8bf*)(p + 8);
  return u.v;
}
// B-frag (32x16 bf16, K x N): lane n holds k = e + half*16 -> 16 contiguous bf16.
static __device__ inline v16bf load_b_frag(const bf16* p) {
  union { v16bf v; v8bf h[2]; } u;
  u.h[0] = *(const v8bf*)(p);
  u.h[1] = *(const v8bf*)(p + 8);
  return u.v;
}
static __device__ inline v8f wmma_bf16(v16bf a, v16bf b, v8f c) {
  return __builtin_amdgcn_wmma_f32_16x16x32_bf16(false, a, false, b, (short)0, c,
                                                 false, false);
}

// ---- weight fp32 -> bf16 conversion (interleaved per-block layout) ---------
// `mul` lets us fold the attention softmax scale into wq at conversion time.
__global__ void convert_blocks_kernel(const float* __restrict__ src,
                                      bf16* __restrict__ dst,
                                      int per, unsigned long long dstStride,
                                      float mul) {
  size_t i = (size_t)blockIdx.x * blockDim.x + threadIdx.x;
  const size_t total = (size_t)NBLK * per;
  const size_t step = (size_t)gridDim.x * blockDim.x;
  for (; i < total; i += step) {
    size_t blk = i / per, r = i - blk * per;
    dst[blk * dstStride + r] = (bf16)(src[i] * mul);
  }
}

// ---- token embedding: x[b,3t+{0,1,2},:] = re/se/ae -------------------------
__global__ __launch_bounds__(256) void embed_kernel(
    const float* __restrict__ R, const float* __restrict__ s,
    const int* __restrict__ a, const int* __restrict__ t,
    const float* __restrict__ pos_emb, const float* __restrict__ act_emb,
    const float* __restrict__ se_w, const float* __restrict__ se_b,
    const float* __restrict__ re_w, const float* __restrict__ re_b,
    float* __restrict__ x) {
  const int bt = blockIdx.x;
  const int b = bt / TT, tt = bt - b * TT;
  const int tid = threadIdx.x;
  __shared__ float sv[OBS];
  if (tid < OBS) sv[tid] = s[(size_t)bt * OBS + tid];
  __syncthreads();
  const int ti = t[bt];
  const int ai = a[bt];
  const float Rv = R[bt];
  float* xr = x + ((size_t)b * SS + 3 * tt) * EE;
  for (int e = tid; e < EE; e += 256) {
    float p = pos_emb[(size_t)ti * EE + e];
    xr[e] = Rv * re_w[e] + re_b[e] + p;
    float accv = se_b[e];
    const float* wr = se_w + (size_t)e * OBS;
#pragma unroll 8
    for (int o = 0; o < OBS; ++o) accv += wr[o] * sv[o];
    xr[EE + e] = accv + p;
    xr[2 * EE + e] = act_emb[(size_t)ai * EE + e] + p;
  }
}

// ---- LayerNorm over E=512 per row (ds_swizzle wave reduction) --------------
template <typename OutT>
__global__ __launch_bounds__(256) void ln_kernel(const float* __restrict__ x,
                                                 const float* __restrict__ g,
                                                 const float* __restrict__ b,
                                                 OutT* __restrict__ out) {
  const int row = blockIdx.x;
  const int tid = threadIdx.x;
  const float* xr = x + (size_t)row * EE;
  float v0 = xr[tid], v1 = xr[tid + 256];
  float sm = v0 + v1, sq = v0 * v0 + v1 * v1;
  sm += swz<SWZ_X1>(sm);  sq += swz<SWZ_X1>(sq);
  sm += swz<SWZ_X2>(sm);  sq += swz<SWZ_X2>(sq);
  sm += swz<SWZ_X4>(sm);  sq += swz<SWZ_X4>(sq);
  sm += swz<SWZ_X8>(sm);  sq += swz<SWZ_X8>(sq);
  sm += swz<SWZ_X16>(sm); sq += swz<SWZ_X16>(sq);
  __shared__ float rs[8], rq[8];
  if ((tid & 31) == 0) { rs[tid >> 5] = sm; rq[tid >> 5] = sq; }
  __syncthreads();
  float ts = 0.f, tq = 0.f;
#pragma unroll
  for (int i = 0; i < 8; ++i) { ts += rs[i]; tq += rq[i]; }
  const float mean = ts * (1.0f / EE);
  const float var = tq * (1.0f / EE) - mean * mean;
  const float inv = rsqrtf(var + 1e-5f);
  out[(size_t)row * EE + tid] = (OutT)((v0 - mean) * inv * g[tid] + b[tid]);
  out[(size_t)row * EE + tid + 256] =
      (OutT)((v1 - mean) * inv * g[tid + 256] + b[tid + 256]);
}

// ---- generic WMMA GEMM: out = A[MxK] * W[NxK]^T ----------------------------
// Wave tile 32x64, register double-buffered k-loop.
// EPI 0: store bf16 (no bias) | 1: +bias, ReLU, store bf16 | 2: +bias, += float out
template <int EPI>
__global__ __launch_bounds__(256) void gemm_wmma_kernel(
    const bf16* __restrict__ A, const bf16* __restrict__ W,
    const float* __restrict__ bias, void* __restrict__ outv, int N, int K) {
  const int lane = threadIdx.x & 31;
  const int wave = threadIdx.x >> 5;
  const int half = lane >> 4;
  const int l15 = lane & 15;
  const int rBase = blockIdx.y * 256 + wave * 32;
  const int cBase = blockIdx.x * 64;

  v8f acc[2][4];
#pragma unroll
  for (int mi = 0; mi < 2; ++mi)
#pragma unroll
    for (int j = 0; j < 4; ++j)
#pragma unroll
      for (int r = 0; r < 8; ++r) acc[mi][j][r] = 0.0f;

  const bf16* aR0 = A + (size_t)(rBase + l15) * K + half * 16;
  const bf16* aR1 = aR0 + (size_t)16 * K;
  const bf16* w0 = W + (size_t)(cBase + l15) * K + half * 16;
  const size_t wT = (size_t)16 * K;

  v16bf aC0 = load_a_frag(aR0);
  v16bf aC1 = load_a_frag(aR1);
  v16bf bC[4];
#pragma unroll
  for (int j = 0; j < 4; ++j) bC[j] = load_b_frag(w0 + j * wT);

  for (int k0 = 32; k0 < K; k0 += 32) {
    v16bf aN0 = load_a_frag(aR0 + k0);
    v16bf aN1 = load_a_frag(aR1 + k0);
    v16bf bN[4];
#pragma unroll
    for (int j = 0; j < 4; ++j) bN[j] = load_b_frag(w0 + j * wT + k0);
    __builtin_prefetch(w0 + k0 + 128, 0, 1);  // -> global_prefetch_b8
#pragma unroll
    for (int j = 0; j < 4; ++j) {
      acc[0][j] = wmma_bf16(aC0, bC[j], acc[0][j]);
      acc[1][j] = wmma_bf16(aC1, bC[j], acc[1][j]);
    }
    aC0 = aN0;
    aC1 = aN1;
#pragma unroll
    for (int j = 0; j < 4; ++j) bC[j] = bN[j];
  }
#pragma unroll
  for (int j = 0; j < 4; ++j) {
    acc[0][j] = wmma_bf16(aC0, bC[j], acc[0][j]);
    acc[1][j] = wmma_bf16(aC1, bC[j], acc[1][j]);
  }

#pragma unroll
  for (int mi = 0; mi < 2; ++mi) {
#pragma unroll
    for (int j = 0; j < 4; ++j) {
      const int col = cBase + j * 16 + l15;
      const float bv = (EPI != 0) ? bias[col] : 0.0f;
#pragma unroll
      for (int r = 0; r < 8; ++r) {
        const int row = rBase + mi * 16 + r + half * 8;
        float v = acc[mi][j][r];
        if (EPI == 0) {
          ((bf16*)outv)[(size_t)row * N + col] = (bf16)v;
        } else if (EPI == 1) {
          v += bv;
          v = v > 0.0f ? v : 0.0f;
          ((bf16*)outv)[(size_t)row * N + col] = (bf16)v;
        } else {
          ((float*)outv)[(size_t)row * N + col] += v + bv;
        }
      }
    }
  }
}

// ---- fused causal flash attention per (b,h) --------------------------------
// Scores computed TRANSPOSED (D[key][query], A=K rows, B=Q rows): softmax over
// keys is an in-lane reduction over accumulator elements + one xor-16 swizzle.
// The 1/sqrt(E) scale is pre-folded into the wq weights, and the causal mask
// is applied only in the (wave-uniform) steps that actually touch the diagonal.
__global__ __launch_bounds__(128) void attn_kernel(const bf16* __restrict__ q,
                                                   const bf16* __restrict__ k,
                                                   const bf16* __restrict__ v,
                                                   bf16* __restrict__ o) {
  const int bh = blockIdx.y;
  const int b = bh >> 2, h = bh & (HH - 1);
  const int qBase = blockIdx.x * 64;
  const int lane = threadIdx.x & 31;
  const int wave = threadIdx.x >> 5;
  const int half = lane >> 4;
  const int l15 = lane & 15;
  const int qTile = qBase + wave * 16;

  const size_t rowOff = (size_t)b * SS;
  const bf16* Qp = q + (rowOff + qTile + l15) * QS + h * HD + half * 16;
  const bf16* Kp = k + rowOff * QS + h * HD;
  const bf16* Vp = v + rowOff * QS + h * HD;

  // Q as B-fragments (lane = query row, contiguous 16 bf16 per k-step)
  v16bf qfB[4];
#pragma unroll
  for (int kk = 0; kk < 4; ++kk) qfB[kk] = load_b_frag(Qp + kk * 32);

  v8f accO[8];
#pragma unroll
  for (int dt = 0; dt < 8; ++dt)
#pragma unroll
    for (int r = 0; r < 8; ++r) accO[dt][r] = 0.0f;
  float Ms = -3.0e38f, Ls = 0.0f;  // per-lane (query = qTile + l15)

  __shared__ bf16 Vt[HD * VTS];
  __shared__ bf16 Pbuf[4][16 * 32];
  bf16* pw = Pbuf[wave];

  const int jEnd = qBase + 64;
  for (int j0 = 0; j0 < jEnd; j0 += 32) {
    __syncthreads();
    // cooperative V^T staging: 8B coalesced global loads, transpose on LDS store
    {
      const int c = threadIdx.x & 31;   // 4-elem d-chunk
      const int kw = threadIdx.x >> 5;  // key row within group of 4
#pragma unroll
      for (int it = 0; it < 8; ++it) {
        const int kk = kw + it * 4;
        v4bf tmp = *(const v4bf*)(Vp + (size_t)(j0 + kk) * QS + c * 4);
#pragma unroll
        for (int u = 0; u < 4; ++u) Vt[(c * 4 + u) * VTS + kk] = tmp[u];
      }
    }
    __syncthreads();

    if (j0 <= qTile + 15) {  // wave-uniform causal skip
      if (j0 + 32 < jEnd)
        __builtin_prefetch(Kp + (size_t)(j0 + 32 + l15) * QS, 0, 1);
      // K tiles as A-fragments (lane = key row)
      v16bf kf[2][4];
#pragma unroll
      for (int c = 0; c < 2; ++c)
#pragma unroll
        for (int kk = 0; kk < 4; ++kk)
          kf[c][kk] = load_a_frag(Kp + (size_t)(j0 + c * 16 + l15) * QS +
                                  kk * 32 + half * 16);
      v8f sc[2];
#pragma unroll
      for (int c = 0; c < 2; ++c)
#pragma unroll
        for (int r = 0; r < 8; ++r) sc[c][r] = 0.0f;
#pragma unroll
      for (int kk = 0; kk < 4; ++kk) {
        sc[0] = wmma_bf16(kf[0][kk], qfB[kk], sc[0]);
        sc[1] = wmma_bf16(kf[1][kk], qfB[kk], sc[1]);
      }
      // causal mask only in diagonal-touching steps (wave-uniform predicate);
      // element key = j0 + c*16 + r + half*8, query = qTile + l15
      if (j0 + 31 > qTile) {
        const int qcol = qTile + l15;
#pragma unroll
        for (int c = 0; c < 2; ++c)
#pragma unroll
          for (int r = 0; r < 8; ++r) {
            const int key = j0 + c * 16 + r + half * 8;
            if (key > qcol) sc[c][r] = -3.0e38f;
          }
      }
      // softmax stats: in-lane over 16 elements + one cross-half swizzle
      float mx = -3.0e38f;
#pragma unroll
      for (int r = 0; r < 8; ++r) mx = fmaxf(mx, fmaxf(sc[0][r], sc[1][r]));
      mx = fmaxf(mx, swz<SWZ_X16>(mx));
      const float Mn = fmaxf(Ms, mx);
      const float alpha = __expf(Ms - Mn);
      Ms = Mn;
#pragma unroll
      for (int c = 0; c < 2; ++c)
#pragma unroll
        for (int r = 0; r < 8; ++r) sc[c][r] = __expf(sc[c][r] - Mn);
      float ps = 0.0f;
#pragma unroll
      for (int r = 0; r < 8; ++r) ps += sc[0][r] + sc[1][r];
      ps += swz<SWZ_X16>(ps);
      Ls = Ls * alpha + ps;
      // broadcast alpha (per query col) to row-indexed v8f, rescale O
      v8f av;
#pragma unroll
      for (int r = 0; r < 8; ++r) av[r] = __shfl(alpha, r + half * 8, 32);
#pragma unroll
      for (int dt = 0; dt < 8; ++dt)
#pragma unroll
        for (int r = 0; r < 8; ++r) accO[dt][r] *= av[r];

      // P^T tile -> LDS in A-layout: per-lane CONTIGUOUS 16B store per tile
#pragma unroll
      for (int c = 0; c < 2; ++c) {
        v8bf pk;
#pragma unroll
        for (int r = 0; r < 8; ++r) pk[r] = (bf16)sc[c][r];
        *(v8bf*)(pw + l15 * 32 + c * 16 + half * 8) = pk;
      }
      asm volatile("s_wait_dscnt 0x0" ::: "memory");
      v16bf pf = load_a_frag(pw + l15 * 32 + half * 16);
#pragma unroll
      for (int g = 0; g < 2; ++g) {
        v16bf vf[4];
#pragma unroll
        for (int d4 = 0; d4 < 4; ++d4)
          vf[d4] = load_b_frag(Vt + ((g * 4 + d4) * 16 + l15) * VTS + half * 16);
#pragma unroll
        for (int d4 = 0; d4 < 4; ++d4)
          accO[g * 4 + d4] = wmma_bf16(pf, vf[d4], accO[g * 4 + d4]);
      }
    }
  }

  const float invL = 1.0f / Ls;
  v8f iv;
#pragma unroll
  for (int r = 0; r < 8; ++r) iv[r] = __shfl(invL, r + half * 8, 32);
  bf16* orow = o + (rowOff + qTile) * EE + h * HD;
#pragma unroll
  for (int dt = 0; dt < 8; ++dt)
#pragma unroll
    for (int r = 0; r < 8; ++r)
      orow[(size_t)(r + half * 8) * EE + dt * 16 + l15] =
          (bf16)(accO[dt][r] * iv[r]);
}

// ---- output heads ----------------------------------------------------------
__global__ __launch_bounds__(256) void heads_kernel(
    const float* __restrict__ xf, const float* __restrict__ linS_w,
    const float* __restrict__ linS_b, const float* __restrict__ linA_w,
    const float* __restrict__ linA_b, const float* __restrict__ linR_w,
    const float* __restrict__ linR_b, float* __restrict__ outR,
    float* __restrict__ outS, float* __restrict__ outA) {
  const int bt = blockIdx.x;
  const int b = bt / TT, tt = bt - b * TT;
  const int tid = threadIdx.x;
  __shared__ float x1[EE], x2[EE];
  const float* r1 = xf + ((size_t)b * SS + 3 * tt + 1) * EE;
  const float* r2 = xf + ((size_t)b * SS + 3 * tt + 2) * EE;
  for (int e = tid; e < EE; e += 256) { x1[e] = r1[e]; x2[e] = r2[e]; }
  __syncthreads();
  if (tid < OBS) {
    const float* w = linS_w + (size_t)tid * EE;
    float acc = linS_b[tid];
    for (int e = 0; e < EE; ++e) acc += w[e] * x2[e];
    outS[(size_t)bt * OBS + tid] = acc;
  } else if (tid < OBS + ACTN) {
    const int j = tid - OBS;
    const float* w = linA_w + (size_t)j * EE;
    float acc = linA_b[j];
    for (int e = 0; e < EE; ++e) acc += w[e] * x1[e];
    outA[(size_t)bt * ACTN + j] = acc;
  } else if (tid == OBS + ACTN) {
    float acc = linR_b[0];
    for (int e = 0; e < EE; ++e) acc += linR_w[e] * x2[e];
    outR[bt] = acc;
  }
}

// ---- launcher --------------------------------------------------------------
extern "C" void kernel_launch(void* const* d_in, const int* in_sizes, int n_in,
                              void* d_out, int out_size, void* d_ws,
                              size_t ws_size, hipStream_t stream) {
  (void)in_sizes; (void)n_in; (void)out_size; (void)ws_size;
  const float* R = (const float*)d_in[0];
  const float* s = (const float*)d_in[1];
  const int* a = (const int*)d_in[2];
  const int* t = (const int*)d_in[3];
  const float* pos_emb = (const float*)d_in[4];
  const float* act_emb = (const float*)d_in[5];
  const float* se_w = (const float*)d_in[6];
  const float* se_b = (const float*)d_in[7];
  const float* re_w = (const float*)d_in[8];
  const float* re_b = (const float*)d_in[9];
  const float* ln1_g = (const float*)d_in[10];
  const float* ln1_b = (const float*)d_in[11];
  const float* wq = (const float*)d_in[12];
  const float* wk = (const float*)d_in[13];
  const float* wv = (const float*)d_in[14];
  const float* proj_w = (const float*)d_in[15];
  const float* proj_b = (const float*)d_in[16];
  const float* ln2_g = (const float*)d_in[17];
  const float* ln2_b = (const float*)d_in[18];
  const float* ff1_w = (const float*)d_in[19];
  const float* ff1_b = (const float*)d_in[20];
  const float* ff2_w = (const float*)d_in[21];
  const float* ff2_b = (const float*)d_in[22];
  const float* lnf_g = (const float*)d_in[23];
  const float* lnf_b = (const float*)d_in[24];
  const float* linS_w = (const float*)d_in[25];
  const float* linS_b = (const float*)d_in[26];
  const float* linA_w = (const float*)d_in[27];
  const float* linA_b = (const float*)d_in[28];
  const float* linR_w = (const float*)d_in[29];
  const float* linR_b = (const float*)d_in[30];

  char* ws = (char*)d_ws;
  size_t off = 0;
  auto take = [&](size_t bytes) {
    char* p = ws + off;
    off += (bytes + 255) & ~(size_t)255;
    return p;
  };
  float* x = (float*)take((size_t)NTOK * EE * 4);    // residual stream (fp32)
  bf16* hb = (bf16*)take((size_t)NTOK * EE * 2);     // LN output (GEMM A)
  bf16* qkv = (bf16*)take((size_t)NTOK * QS * 2);    // packed Q|K|V rows
  bf16* ob = (bf16*)take((size_t)NTOK * EE * 2);     // attention output
  bf16* fb = (bf16*)take((size_t)NTOK * FFD * 2);    // FF hidden
  const size_t PB = 4 * (size_t)EE * EE + 2 * (size_t)FFD * EE;  // per-block wts
  bf16* wbf = (bf16*)take(NBLK * PB * 2);
  float* xf = (float*)fb;  // final-LN output reuses FF buffer (consumed by then)

  // one-shot weight conversion fp32 -> bf16 (wq|wk|wv contiguous per block);
  // 1/sqrt(E) softmax scale folded into wq.
  {
    const float qscale = 0.04419417382415922f;
    dim3 g(1024), blk(256);
    convert_blocks_kernel<<<g, blk, 0, stream>>>(wq, wbf, EE * EE, PB, qscale);
    convert_blocks_kernel<<<g, blk, 0, stream>>>(wk, wbf + (size_t)EE * EE, EE * EE, PB, 1.0f);
    convert_blocks_kernel<<<g, blk, 0, stream>>>(wv, wbf + 2 * (size_t)EE * EE, EE * EE, PB, 1.0f);
    convert_blocks_kernel<<<g, blk, 0, stream>>>(proj_w, wbf + 3 * (size_t)EE * EE, EE * EE, PB, 1.0f);
    convert_blocks_kernel<<<g, blk, 0, stream>>>(ff1_w, wbf + 4 * (size_t)EE * EE, FFD * EE, PB, 1.0f);
    convert_blocks_kernel<<<g, blk, 0, stream>>>(
        ff2_w, wbf + 4 * (size_t)EE * EE + (size_t)FFD * EE, EE * FFD, PB, 1.0f);
  }

  embed_kernel<<<BB * TT, 256, 0, stream>>>(R, s, a, t, pos_emb, act_emb, se_w,
                                            se_b, re_w, re_b, x);

  const dim3 gE(EE / 64, NTOK / 256);
  const dim3 gQKV(QS / 64, NTOK / 256);
  const dim3 gF(FFD / 64, NTOK / 256);
  for (int i = 0; i < NBLK; ++i) {
    const bf16* wqkv = wbf + (size_t)i * PB;              // [3E][E] fused
    const bf16* wpi = wqkv + 3 * (size_t)EE * EE;
    const bf16* wf1 = wpi + (size_t)EE * EE;
    const bf16* wf2 = wf1 + (size_t)FFD * EE;

    ln_kernel<bf16><<<NTOK, 256, 0, stream>>>(x, ln1_g + i * EE, ln1_b + i * EE, hb);
    gemm_wmma_kernel<0><<<gQKV, 256, 0, stream>>>(hb, wqkv, nullptr, qkv, QS, EE);
    attn_kernel<<<dim3(SS / 64, BB * HH), 128, 0, stream>>>(
        qkv, qkv + EE, qkv + 2 * EE, ob);
    gemm_wmma_kernel<2><<<gE, 256, 0, stream>>>(ob, wpi, proj_b + i * EE, x, EE, EE);
    ln_kernel<bf16><<<NTOK, 256, 0, stream>>>(x, ln2_g + i * EE, ln2_b + i * EE, hb);
    gemm_wmma_kernel<1><<<gF, 256, 0, stream>>>(hb, wf1, ff1_b + i * FFD, fb, FFD, EE);
    gemm_wmma_kernel<2><<<gE, 256, 0, stream>>>(fb, wf2, ff2_b + i * EE, x, EE, FFD);
  }

  ln_kernel<float><<<NTOK, 256, 0, stream>>>(x, lnf_g, lnf_b, xf);

  float* outR = (float*)d_out;
  float* outS = outR + (size_t)BB * TT;
  float* outA = outS + (size_t)BB * TT * OBS;
  heads_kernel<<<BB * TT, 256, 0, stream>>>(xf, linS_w, linS_b, linA_w, linA_b,
                                            linR_w, linR_b, outR, outS, outA);
}
